// PseudoTokenInitialiser_49374944034931
// MI455X (gfx1250) — matrix operand order repaired and checked by reading, not compiled, not run on hardware
//
#include <hip/hip_runtime.h>
#include <hip/hip_bf16.h>

#define M_B   8
#define NQ    256
#define NKV   4096
#define E_DIM 1024
#define H_N   16
#define D_H   64
#define HD    1024
#define QK_SCALE 0.125f   // 64^-0.5

typedef __bf16 bf16;
typedef __attribute__((ext_vector_type(16))) __bf16 v16bf;
typedef __attribute__((ext_vector_type(8)))  __bf16 v8bf;
typedef __attribute__((ext_vector_type(8)))  float  v8f;
typedef __attribute__((ext_vector_type(4)))  float  v4f;

// Build a 16-element bf16 fragment from two contiguous 8-element (16B) chunks.
static __device__ __forceinline__ v16bf make_frag(const bf16* lo, const bf16* hi) {
  v8bf a = *(const v8bf*)lo;
  v8bf b = *(const v8bf*)hi;
  v16bf r;
#pragma unroll
  for (int i = 0; i < 8; ++i) { r[i] = a[i]; r[i + 8] = b[i]; }
  return r;
}

static __device__ __forceinline__ v8f wmma_bf16(v16bf a, v16bf b, v8f c) {
  return __builtin_amdgcn_wmma_f32_16x16x32_bf16(false, a, false, b, (short)0, c,
                                                 false, false);
}

// ---------------------------------------------------------------------------
// Kernel 0: tiled transpose + f32->bf16 convert:  out[C][R] = (bf16) in[R][C]
// Gives every GEMM B-operand a [n][k] layout (K-dim contiguous) so WMMA
// B-fragments become two contiguous 16B global loads (no LDS staging later).
// ---------------------------------------------------------------------------
__global__ __launch_bounds__(256)
void trans_cvt(const float* __restrict__ in, bf16* __restrict__ out,
               int R, int C, long inZ, long outZ) {
  __shared__ bf16 T[64][80];  // 64x64 tile, padded stride 160B (16B aligned)
  const int tid = threadIdx.x;
  const int r0 = blockIdx.y * 64;
  const int c0 = blockIdx.x * 64;
  const float* ip = in + (size_t)blockIdx.z * inZ;
  bf16* op = out + (size_t)blockIdx.z * outZ;
  {
    const int rr = tid >> 2;          // 0..63
    const int cc = (tid & 3) * 16;    // 0,16,32,48
    const float* src = ip + (size_t)(r0 + rr) * C + c0 + cc;
#pragma unroll
    for (int j = 0; j < 16; ++j) T[cc + j][rr] = (bf16)src[j];
  }
  __syncthreads();
  {
    const int rr = tid >> 2;          // row of transposed tile (orig col)
    const int cc = (tid & 3) * 16;
    bf16* dst = op + (size_t)(c0 + rr) * R + r0 + cc;
    v8bf a, b;
#pragma unroll
    for (int j = 0; j < 8; ++j) { a[j] = T[rr][cc + j]; b[j] = T[rr][cc + 8 + j]; }
    *(v8bf*)dst = a;
    *(v8bf*)(dst + 8) = b;
  }
}

// ---------------------------------------------------------------------------
// Kernel 1: P[rows,1024](bf16) = Z[rows,1024](f32) @ W  with W given as
// WT[n][k] bf16.  A tile staged f32->bf16 through LDS; B-fragments loaded
// straight from global (WT is 2MB -> L2 resident).  C-tile 128x128,
// wave tile 32x64, BK=32, 8 WMMA per staging step.
// grid.x = col tile (fast-varying so concurrent blocks share the Z row slice
// in L2), grid.y = row tile.
// ---------------------------------------------------------------------------
__global__ __launch_bounds__(256)
void proj_gemm(const float* __restrict__ Z, const bf16* __restrict__ WT,
               bf16* __restrict__ P) {
  __shared__ bf16 As[128][48];  // [row][k], stride 96B
  const int tid    = threadIdx.x;
  const int wave   = tid >> 5;
  const int lane   = tid & 31;
  const int lr     = lane & 15;
  const int laneHi = lane >> 4;
  const int colBase = blockIdx.x * 128;
  const int rowBase = blockIdx.y * 128;
  const int wr = wave >> 1;  // 0..3
  const int wc = wave & 1;   // 0..1
  const int lo = laneHi ? 8 : 0;
  const int ko = laneHi ? 8 : 0;

  v8f acc[2][4];
#pragma unroll
  for (int i = 0; i < 2; ++i)
#pragma unroll
    for (int j = 0; j < 4; ++j) acc[i][j] = (v8f){};

  const bf16* bptr[4];
#pragma unroll
  for (int j = 0; j < 4; ++j)
    bptr[j] = WT + (size_t)(colBase + wc * 64 + j * 16 + lr) * E_DIM + ko;

  for (int k0 = 0; k0 < E_DIM; k0 += 32) {
    {  // stage A: 128x32 f32 -> bf16
      const int c4 = (tid & 7) * 4;
      const int r0 = tid >> 3;
#pragma unroll
      for (int it = 0; it < 4; ++it) {
        const int row = r0 + it * 32;
        v4f v = *(const v4f*)(Z + (size_t)(rowBase + row) * E_DIM + k0 + c4);
#pragma unroll
        for (int j = 0; j < 4; ++j) As[row][c4 + j] = (bf16)v[j];
      }
    }
    __syncthreads();
    __builtin_prefetch(bptr[0] + k0 + 32, 0, 0);
    __builtin_prefetch(bptr[2] + k0 + 32, 0, 0);

    v16bf af[2], bfr[4];
#pragma unroll
    for (int i = 0; i < 2; ++i) {
      const int row = wr * 32 + i * 16 + lr;
      af[i] = make_frag(&As[row][lo], &As[row][lo + 16]);
    }
#pragma unroll
    for (int j = 0; j < 4; ++j) bfr[j] = make_frag(bptr[j] + k0, bptr[j] + k0 + 16);
#pragma unroll
    for (int i = 0; i < 2; ++i)
#pragma unroll
      for (int j = 0; j < 4; ++j) acc[i][j] = wmma_bf16(af[i], bfr[j], acc[i][j]);
    __syncthreads();
  }

#pragma unroll
  for (int i = 0; i < 2; ++i)
#pragma unroll
    for (int j = 0; j < 4; ++j) {
      const int col = colBase + wc * 64 + j * 16 + lr;
#pragma unroll
      for (int e = 0; e < 8; ++e) {
        const int row = rowBase + wr * 32 + i * 16 + e + laneHi * 8;
        P[(size_t)row * HD + col] = (bf16)acc[i][j][e];
      }
    }
}

// ---------------------------------------------------------------------------
// Kernel 2a: per-(m,h,16-query-tile) softmax stats via online max/sum.
// ---------------------------------------------------------------------------
__global__ __launch_bounds__(256)
void attn_stats(const bf16* __restrict__ Qb, const bf16* __restrict__ Kb,
                float* __restrict__ rowmax, float* __restrict__ rowsum) {
  __shared__ float lmx[16][128];
  __shared__ float lsm[16][128];
  const int tid = threadIdx.x;
  const int wave = tid >> 5, lane = tid & 31, lr = lane & 15, laneHi = lane >> 4;
  const int qt = blockIdx.x, h = blockIdx.y, m = blockIdx.z;
  const int lo = laneHi ? 8 : 0;

  const bf16* qbase = Qb + ((size_t)(m * NQ + qt * 16 + lr) * HD + h * D_H);
  const v16bf qa0 = make_frag(qbase + lo, qbase + lo + 16);            // d 0..31
  const v16bf qa1 = make_frag(qbase + 32 + lo, qbase + 32 + lo + 16);  // d 32..63

  float mx[8], sm[8];
#pragma unroll
  for (int j = 0; j < 8; ++j) { mx[j] = -__builtin_inff(); sm[j] = 0.f; }

  const int kvStart = wave * (NKV / 8);
  for (int nt = 0; nt < (NKV / 8) / 16; ++nt) {
    const int kv = kvStart + nt * 16 + lr;
    const bf16* kb = Kb + ((size_t)(m * NKV + kv) * HD + h * D_H + (laneHi ? 8 : 0));
    v16bf b0 = make_frag(kb, kb + 16);
    v16bf b1 = make_frag(kb + 32, kb + 48);
    v8f c = (v8f){};
    c = wmma_bf16(qa0, b0, c);
    c = wmma_bf16(qa1, b1, c);
#pragma unroll
    for (int j = 0; j < 8; ++j) {
      const float s = c[j] * QK_SCALE;
      const float nm = fmaxf(mx[j], s);
      sm[j] = sm[j] * __expf(mx[j] - nm) + __expf(s - nm);
      mx[j] = nm;
    }
  }
  const int slot = wave * 16 + lr;
#pragma unroll
  for (int j = 0; j < 8; ++j) {
    const int r = j + laneHi * 8;
    lmx[r][slot] = mx[j];
    lsm[r][slot] = sm[j];
  }
  __syncthreads();
  if (tid < 16) {
    float Mv = -__builtin_inff(), Sv = 0.f;
    for (int i = 0; i < 128; ++i) {
      const float m2 = lmx[tid][i], s2 = lsm[tid][i];
      const float nm = fmaxf(Mv, m2);
      Sv = Sv * __expf(Mv - nm) + s2 * __expf(m2 - nm);
      Mv = nm;
    }
    const size_t idx = ((size_t)(m * H_N + h) * NQ) + qt * 16 + tid;
    rowmax[idx] = Mv;
    rowsum[idx] = Sv;
  }
}

// ---------------------------------------------------------------------------
// Kernel 2b: head-combined attention matrix
//   A[m,q,kv] = sum_h hw[h] * exp(scale*s - max)/sum   (stored bf16)
// 16q x 128kv tile per block; stats + head weights staged in LDS.
// ---------------------------------------------------------------------------
__global__ __launch_bounds__(256)
void attn_combine(const bf16* __restrict__ Qb, const bf16* __restrict__ Kb,
                  const float* __restrict__ rowmax, const float* __restrict__ rowsum,
                  const float* __restrict__ rawhw, bf16* __restrict__ A) {
  __shared__ float hw[16];
  __shared__ float smax[16][16];  // [h][row]
  __shared__ float ssum[16][16];
  const int tid = threadIdx.x;
  const int wave = tid >> 5, lane = tid & 31, lr = lane & 15, laneHi = lane >> 4;
  const int kt = blockIdx.x, qt = blockIdx.y, m = blockIdx.z;
  const int lo = laneHi ? 8 : 0;

  {  // stage per-head stats for this 16-row tile
    const int h = tid >> 4, r = tid & 15;
    const size_t idx = ((size_t)(m * H_N + h) * NQ) + qt * 16 + r;
    smax[h][r] = rowmax[idx];
    ssum[h][r] = rowsum[idx];
  }
  if (tid < 16) {  // softmax of 16 head weights
    float mxv = -__builtin_inff();
    for (int i = 0; i < 16; ++i) mxv = fmaxf(mxv, rawhw[i]);
    float s = 0.f;
    for (int i = 0; i < 16; ++i) s += __expf(rawhw[i] - mxv);
    hw[tid] = __expf(rawhw[tid] - mxv) / s;
  }
  __syncthreads();

  const int kv = kt * 128 + wave * 16 + lr;
  float acc[8];
#pragma unroll
  for (int j = 0; j < 8; ++j) acc[j] = 0.f;

  for (int h = 0; h < H_N; ++h) {
    const bf16* qbase = Qb + ((size_t)(m * NQ + qt * 16 + lr) * HD + h * D_H);
    const v16bf qa0 = make_frag(qbase + lo, qbase + lo + 16);
    const v16bf qa1 = make_frag(qbase + 32 + lo, qbase + 32 + lo + 16);
    const bf16* kb = Kb + ((size_t)(m * NKV + kv) * HD + h * D_H + (laneHi ? 8 : 0));
    v16bf b0 = make_frag(kb, kb + 16);
    v16bf b1 = make_frag(kb + 32, kb + 48);
    v8f c = (v8f){};
    c = wmma_bf16(qa0, b0, c);
    c = wmma_bf16(qa1, b1, c);
    const float wh = hw[h];
#pragma unroll
    for (int j = 0; j < 8; ++j) {
      const int r = j + laneHi * 8;
      acc[j] += wh * __expf(c[j] * QK_SCALE - smax[h][r]) / ssum[h][r];
    }
  }
#pragma unroll
  for (int j = 0; j < 8; ++j) {
    const int r = qt * 16 + j + laneHi * 8;
    A[(size_t)(m * NQ + r) * NKV + kv] = (bf16)acc[j];
  }
}

// ---------------------------------------------------------------------------
// Kernel 3: xq_out = xq + A @ xkv  with xkv given pre-transposed bf16
// xkvT[m][n][k].  LDS-free: all fragments are contiguous global b128 loads.
// C-tile 128q x 128n, wave tile 32x64, BK=32, 8 WMMA per K-step, no barriers.
// grid.x = q tile (fast) so both q tiles sharing an xkvT slice run together.
// ---------------------------------------------------------------------------
__global__ __launch_bounds__(256)
void out_gemm(const bf16* __restrict__ Am, const bf16* __restrict__ xkvT,
              const float* __restrict__ xq, float* __restrict__ out) {
  const int tid = threadIdx.x;
  const int wave = tid >> 5, lane = tid & 31, lr = lane & 15, laneHi = lane >> 4;
  const int rowBase = blockIdx.x * 128;       // q tile (NQ/128 = 2)
  const int colBase = blockIdx.y * 128;       // col tile (E/128 = 8)
  const int m = blockIdx.z;
  const int wr = wave >> 1;
  const int wc = wave & 1;
  const int lo = laneHi ? 8 : 0;
  const int ko = laneHi ? 8 : 0;

  v8f acc[2][4];
#pragma unroll
  for (int i = 0; i < 2; ++i)
#pragma unroll
    for (int j = 0; j < 4; ++j) acc[i][j] = (v8f){};

  const bf16* aptr[2];
  const bf16* bptr[4];
#pragma unroll
  for (int i = 0; i < 2; ++i)
    aptr[i] = Am + (size_t)(m * NQ + rowBase + wr * 32 + i * 16 + lr) * NKV + lo;
#pragma unroll
  for (int j = 0; j < 4; ++j)
    bptr[j] = xkvT + (size_t)(m * E_DIM + colBase + wc * 64 + j * 16 + lr) * NKV + ko;

  for (int k0 = 0; k0 < NKV; k0 += 32) {
    __builtin_prefetch(aptr[0] + k0 + 64, 0, 0);
    __builtin_prefetch(bptr[0] + k0 + 64, 0, 0);
    __builtin_prefetch(bptr[2] + k0 + 64, 0, 0);
    v16bf af[2], bfr[4];
#pragma unroll
    for (int i = 0; i < 2; ++i) af[i] = make_frag(aptr[i] + k0, aptr[i] + k0 + 16);
#pragma unroll
    for (int j = 0; j < 4; ++j) bfr[j] = make_frag(bptr[j] + k0, bptr[j] + k0 + 16);
#pragma unroll
    for (int i = 0; i < 2; ++i)
#pragma unroll
      for (int j = 0; j < 4; ++j) acc[i][j] = wmma_bf16(af[i], bfr[j], acc[i][j]);
  }

#pragma unroll
  for (int i = 0; i < 2; ++i)
#pragma unroll
    for (int j = 0; j < 4; ++j) {
      const int col = colBase + wc * 64 + j * 16 + lr;
#pragma unroll
      for (int e = 0; e < 8; ++e) {
        const int row = rowBase + wr * 32 + i * 16 + e + laneHi * 8;
        const size_t idx = (size_t)(m * NQ + row) * E_DIM + col;
        out[idx] = xq[idx] + acc[i][j][e];
      }
    }
}

// ---------------------------------------------------------------------------
extern "C" void kernel_launch(void* const* d_in, const int* in_sizes, int n_in,
                              void* d_out, int out_size, void* d_ws, size_t ws_size,
                              hipStream_t stream) {
  (void)in_sizes; (void)n_in; (void)out_size; (void)ws_size;
  const float* zq  = (const float*)d_in[0];
  const float* zkv = (const float*)d_in[1];
  const float* xq  = (const float*)d_in[2];
  const float* xkv = (const float*)d_in[3];
  const float* Wq  = (const float*)d_in[4];
  const float* Wk  = (const float*)d_in[5];
  const float* rhw = (const float*)d_in[6];

  const size_t zq_elems = (size_t)M_B * NQ * E_DIM;
  float* out_zq = (float*)d_out;
  float* out_xq = out_zq + zq_elems;

  // Workspace layout (bytes):
  //  WqT 2MB | WkT 2MB | xkvT 64MB | Qb 4MB | Kb 64MB | rmax 128K | rsum 128K | A 16MB
  char* ws = (char*)d_ws;
  bf16*  WqT  = (bf16*)(ws);
  bf16*  WkT  = (bf16*)(ws + (2ull << 20));
  bf16*  xkvT = (bf16*)(ws + (4ull << 20));
  bf16*  Qb   = (bf16*)(ws + (68ull << 20));
  bf16*  Kb   = (bf16*)(ws + (72ull << 20));
  float* rmax = (float*)(ws + (136ull << 20));
  float* rsum = (float*)(ws + (136ull << 20) + (256ull << 10));
  bf16*  Am   = (bf16*)(ws + (137ull << 20));

  // Output 0: zq passthrough
  hipMemcpyAsync(out_zq, d_in[0], zq_elems * sizeof(float),
                 hipMemcpyDeviceToDevice, stream);

  // B-operand pre-transposes (f32 -> bf16, [n][k] layout)
  trans_cvt<<<dim3(HD / 64, E_DIM / 64, 1), 256, 0, stream>>>(
      Wq, WqT, E_DIM, HD, 0, 0);
  trans_cvt<<<dim3(HD / 64, E_DIM / 64, 1), 256, 0, stream>>>(
      Wk, WkT, E_DIM, HD, 0, 0);
  trans_cvt<<<dim3(E_DIM / 64, NKV / 64, M_B), 256, 0, stream>>>(
      xkv, xkvT, NKV, E_DIM, (long)NKV * E_DIM, (long)E_DIM * NKV);

  // Q/K projections (bf16 outputs)
  proj_gemm<<<dim3(HD / 128, (M_B * NQ) / 128), 256, 0, stream>>>(zq, WqT, Qb);
  proj_gemm<<<dim3(HD / 128, (M_B * NKV) / 128), 256, 0, stream>>>(zkv, WkT, Kb);

  // Softmax stats per (m, h, 16-query tile)
  attn_stats<<<dim3(NQ / 16, H_N, M_B), 256, 0, stream>>>(Qb, Kb, rmax, rsum);

  // Head-combined attention matrix A (bf16)
  attn_combine<<<dim3(NKV / 128, NQ / 16, M_B), 256, 0, stream>>>(Qb, Kb, rmax,
                                                                  rsum, rhw, Am);

  // Output 1: xq + A @ xkv
  out_gemm<<<dim3(NQ / 128, E_DIM / 128, M_B), 256, 0, stream>>>(Am, xkvT, xq,
                                                                 out_xq);
}
